// C2FTransformerV13_65403761983784
// MI455X (gfx1250) — compile-verified, matrix operand order checked by reading, not run on hardware
//
#include <hip/hip_runtime.h>
#include <math.h>

// ---------------- problem constants (from reference) ----------------
#define B_ 2
#define S_ 512
#define D_ 1280
#define H_ 20
#define L_ 16
#define DK_ 64
#define LATC_ 112
#define VOCAB_ 1024
#define NPRED_ 10
#define DI_ 2560          // D_INNER
#define W1O_ 5120         // 2*D_INNER
#define TOK_ (B_*S_)      // 1024

// ---------------- GEMM tiling ----------------
// All GEMMs launched from host have M % 128 == 0 (512,1024), N % 64 == 0,
// K % 32 == 0 -> no bounds checks anywhere in the kernel.
#define BM 128
#define BN 64
#define BK 32
#define LDS_STRIDE 36     // halves per row; even and row stride 72B (8B aligned)

typedef __bf16 bf16_t;
typedef bf16_t v16bf __attribute__((ext_vector_type(16)));
typedef float  v8f   __attribute__((ext_vector_type(8)));

__device__ __forceinline__ unsigned short f32_to_bf16(float f) {
  unsigned u = __float_as_uint(f);
  unsigned r = u + 0x7FFFu + ((u >> 16) & 1u);   // round-to-nearest-even
  return (unsigned short)(r >> 16);
}

__device__ __forceinline__ unsigned pack_bf16x2(float a, float b) {
  return (unsigned)f32_to_bf16(a) | ((unsigned)f32_to_bf16(b) << 16);
}

struct GemmParams {
  int M, N, K;
  const float* A; int lda; long sAo, sAi;
  const float* Bm; int ldb; long sBo, sBi; int bT;
  float*       C;  int ldc; long sCo, sCi;
  const float* bias;     // length N (indexed by global col), or null
  const float* resid;    // same layout/offsets as C, or null
  float alpha;
  int zdiv;              // z -> (zo = z/zdiv, zi = z%zdiv)
  int cls_mode;          // classifier permuted output
  int cls_snp;           // S*NPRED
};

// ---------------- core WMMA GEMM (f32 in -> bf16 tiles -> f32 out) ----------------
// 256 threads = 8 wave32, wave grid 4(M) x 2(N), each wave 2x2 WMMA 16x16 tiles.
// Double-buffered LDS ping-pong; vectorized b128 global loads; no bounds checks.
__global__ __launch_bounds__(256) void gemm_bf16_wmma(GemmParams p) {
  __shared__ unsigned short As[2][BM * LDS_STRIDE];
  __shared__ unsigned short Bs[2][BN * LDS_STRIDE];

  const int tid  = threadIdx.x;
  const int lane = tid & 31;
  const int wid  = tid >> 5;        // 0..7
  const int wm   = wid >> 1;        // 0..3
  const int wn   = wid & 1;         // 0..1

  const int bm0 = blockIdx.y * BM;
  const int bn0 = blockIdx.x * BN;
  const int z   = blockIdx.z;
  const int zo  = z / p.zdiv;
  const int zi  = z % p.zdiv;

  const float* A  = p.A  + zo * p.sAo + zi * p.sAi;
  const float* Bp = p.Bm + zo * p.sBo + zi * p.sBi;
  float*       C  = p.C  + zo * p.sCo + zi * p.sCi;
  const float* R  = p.resid ? (p.resid + zo * p.sCo + zi * p.sCi) : nullptr;

  v8f acc[2][2] = {};

  const int nsteps = p.K / BK;

  // ---- tile loaders (exact tiles, vectorized) ----
  auto loadA = [&](int buf, int k0) {
#pragma unroll
    for (int i = 0; i < 4; ++i) {
      int r = (tid >> 3) + i * 32;          // 128 rows
      int c = (tid & 7) * 4;                // 32 cols in float4 chunks
      float4 v = *(const float4*)&A[(long)(bm0 + r) * p.lda + (k0 + c)];
      uint2 pk;
      pk.x = pack_bf16x2(v.x, v.y);
      pk.y = pack_bf16x2(v.z, v.w);
      *(uint2*)&As[buf][r * LDS_STRIDE + c] = pk;
    }
  };
  auto loadB = [&](int buf, int k0) {
    if (p.bT) {
      // B is N x K row-major: direct vectorized copy into Bs[n][k]
#pragma unroll
      for (int i = 0; i < 2; ++i) {
        int n = (tid >> 3) + i * 32;        // 64 rows
        int c = (tid & 7) * 4;
        float4 v = *(const float4*)&Bp[(long)(bn0 + n) * p.ldb + (k0 + c)];
        uint2 pk;
        pk.x = pack_bf16x2(v.x, v.y);
        pk.y = pack_bf16x2(v.z, v.w);
        *(uint2*)&Bs[buf][n * LDS_STRIDE + c] = pk;
      }
    } else {
      // B is K x N row-major: vectorized read along n, transpose into Bs[n][k]
#pragma unroll
      for (int i = 0; i < 2; ++i) {
        int idx = tid + i * 256;            // 0..511
        int k  = idx >> 4;                  // 32 k-rows
        int n0 = (idx & 15) * 4;            // 64 cols in float4 chunks
        float4 v = *(const float4*)&Bp[(long)(k0 + k) * p.ldb + (bn0 + n0)];
        Bs[buf][(n0 + 0) * LDS_STRIDE + k] = f32_to_bf16(v.x);
        Bs[buf][(n0 + 1) * LDS_STRIDE + k] = f32_to_bf16(v.y);
        Bs[buf][(n0 + 2) * LDS_STRIDE + k] = f32_to_bf16(v.z);
        Bs[buf][(n0 + 3) * LDS_STRIDE + k] = f32_to_bf16(v.w);
      }
    }
  };

  loadA(0, 0);
  loadB(0, 0);
  __syncthreads();

  const int r15 = lane & 15;
  const int kh  = (lane >> 4) * 8;

  for (int s = 0; s < nsteps; ++s) {
    const int cur = s & 1;

    if (s + 1 < nsteps) {
      if (s + 2 < nsteps) {   // prefetch tile after next (global_prefetch_b8)
        int pr = bm0 + (tid >> 1);
        int pc = (s + 2) * BK + ((tid & 1) << 4);
        __builtin_prefetch(&A[(long)pr * p.lda + pc], 0, 0);
      }
      loadA(cur ^ 1, (s + 1) * BK);
      loadB(cur ^ 1, (s + 1) * BK);
    }

    // ---- fragments per 16-bit A 16x32 lane layout ----
    union Frag { unsigned u[8]; v16bf v; };
    Frag fa[2], fb[2];
#pragma unroll
    for (int vv = 0; vv < 8; ++vv) {
      int kk = (vv < 4 ? 2 * vv : 16 + 2 * (vv - 4)) + kh;
#pragma unroll
      for (int mi = 0; mi < 2; ++mi)
        fa[mi].u[vv] = *(const unsigned*)&As[cur][(wm * 32 + mi * 16 + r15) * LDS_STRIDE + kk];
#pragma unroll
      for (int ni = 0; ni < 2; ++ni)
        fb[ni].u[vv] = *(const unsigned*)&Bs[cur][(wn * 32 + ni * 16 + r15) * LDS_STRIDE + kk];
    }

#pragma unroll
    for (int mi = 0; mi < 2; ++mi)
#pragma unroll
      for (int ni = 0; ni < 2; ++ni)
        acc[mi][ni] = __builtin_amdgcn_wmma_f32_16x16x32_bf16(
            false, fa[mi].v, false, fb[ni].v, (short)0, acc[mi][ni], false, false);

    __syncthreads();
  }

  // ---- epilogue: lanes0-15: VGPR r -> M=r; lanes16-31: M=r+8 (no bounds checks) ----
  const int rowAdd = (lane >> 4) ? 8 : 0;
#pragma unroll
  for (int mi = 0; mi < 2; ++mi) {
#pragma unroll
    for (int ni = 0; ni < 2; ++ni) {
      int col = bn0 + wn * 32 + ni * 16 + (lane & 15);
      float bi = p.bias ? p.bias[col] : 0.0f;
#pragma unroll
      for (int r = 0; r < 8; ++r) {
        int row = bm0 + wm * 32 + mi * 16 + r + rowAdd;
        float v = acc[mi][ni][r] * p.alpha + bi;
        if (R) v += R[(long)row * p.ldc + col];
        if (p.cls_mode) {
          int vc = col / NPRED_, pp = col % NPRED_;
          C[(long)vc * p.cls_snp + (long)row * NPRED_ + pp] = v;
        } else {
          C[(long)row * p.ldc + col] = v;
        }
      }
    }
  }
}

// ---------------- embedding: x[b,n,d] = sum_c lat[b,c,n]*emb_w[d,c] + emb_b[d] ----------------
__global__ __launch_bounds__(256) void embed_kernel(const float* __restrict__ lat,
                                                    const float* __restrict__ ew,
                                                    const float* __restrict__ eb,
                                                    float* __restrict__ x) {
  long idx = (long)blockIdx.x * blockDim.x + threadIdx.x;
  if (idx >= (long)TOK_ * D_) return;
  int d = (int)(idx % D_);
  int t = (int)(idx / D_);
  int b = t / S_, n = t % S_;
  const float* lp = lat + (long)b * LATC_ * S_ + n;
  const float* wp = ew + (long)d * LATC_;
  float s = eb[d];
#pragma unroll 4
  for (int c = 0; c < LATC_; ++c) s += lp[(long)c * S_] * wp[c];
  x[idx] = s;
}

// ---------------- rmsnorm ----------------
__global__ __launch_bounds__(256) void rmsnorm_kernel(const float* __restrict__ x,
                                                      const float* __restrict__ w,
                                                      float* __restrict__ out) {
  __shared__ float red[256];
  int row = blockIdx.x;
  const float* xr = x + (long)row * D_;
  float s = 0.0f;
  for (int i = threadIdx.x; i < D_; i += 256) { float v = xr[i]; s += v * v; }
  red[threadIdx.x] = s; __syncthreads();
  for (int off = 128; off > 0; off >>= 1) {
    if (threadIdx.x < off) red[threadIdx.x] += red[threadIdx.x + off];
    __syncthreads();
  }
  float scale = rsqrtf(red[0] / (float)D_ + 1e-5f);
  for (int i = threadIdx.x; i < D_; i += 256)
    out[(long)row * D_ + i] = w[i] * xr[i] * scale;
}

// ---------------- softmax over S with optional T5 relative bias (layer 0) ----------------
__global__ __launch_bounds__(256) void softmax_kernel(float* __restrict__ scores,
                                                      const float* __restrict__ rel_bias,
                                                      int add_bias) {
  __shared__ float red[256];
  long rid = blockIdx.x;               // over B*H*S rows
  int i  = (int)(rid % S_);
  int bh = (int)(rid / S_);
  int h  = bh % H_;
  float* row = scores + rid * S_;

  float vals[2];
  float mx = -1e30f;
#pragma unroll
  for (int t = 0; t < 2; ++t) {
    int j = threadIdx.x + t * 256;
    float v = row[j];
    if (add_bias) {
      int nrel = i - j;
      int ret = (nrel < 0) ? 16 : 0;
      int n = nrel < 0 ? -nrel : nrel;
      int bucket;
      if (n < 8) bucket = ret + n;
      else {
        int vl = 8 + (int)(logf((float)n * 0.125f) * (8.0f / logf(16.0f)));
        if (vl > 15) vl = 15;
        bucket = ret + vl;
      }
      v += rel_bias[bucket * H_ + h];
    }
    vals[t] = v;
    mx = fmaxf(mx, v);
  }
  red[threadIdx.x] = mx; __syncthreads();
  for (int off = 128; off > 0; off >>= 1) {
    if (threadIdx.x < off) red[threadIdx.x] = fmaxf(red[threadIdx.x], red[threadIdx.x + off]);
    __syncthreads();
  }
  mx = red[0]; __syncthreads();
  float sum = 0.0f;
#pragma unroll
  for (int t = 0; t < 2; ++t) { vals[t] = __expf(vals[t] - mx); sum += vals[t]; }
  red[threadIdx.x] = sum; __syncthreads();
  for (int off = 128; off > 0; off >>= 1) {
    if (threadIdx.x < off) red[threadIdx.x] += red[threadIdx.x + off];
    __syncthreads();
  }
  float inv = 1.0f / red[0];
#pragma unroll
  for (int t = 0; t < 2; ++t) row[threadIdx.x + t * 256] = vals[t] * inv;
}

// ---------------- gated GELU: f = a * gelu(g) ----------------
__global__ __launch_bounds__(256) void act_kernel(const float* __restrict__ u,
                                                  float* __restrict__ f) {
  long idx = (long)blockIdx.x * blockDim.x + threadIdx.x;
  if (idx >= (long)TOK_ * DI_) return;
  long row = idx / DI_;
  long j   = idx % DI_;
  float a = u[row * W1O_ + j];
  float g = u[row * W1O_ + DI_ + j];
  float g3 = g + 0.044715f * g * g * g;
  float t  = tanhf(0.7978845608028654f * g3);
  f[idx] = a * 0.5f * g * (1.0f + t);
}

// ---------------- host orchestration ----------------
static inline void run_gemm(const GemmParams& p, int batch, hipStream_t s) {
  dim3 grid((p.N + BN - 1) / BN, (p.M + BM - 1) / BM, batch);
  gemm_bf16_wmma<<<grid, 256, 0, s>>>(p);
}

extern "C" void kernel_launch(void* const* d_in, const int* in_sizes, int n_in,
                              void* d_out, int out_size, void* d_ws, size_t ws_size,
                              hipStream_t stream) {
  const float* latents = (const float*)d_in[0];
  const float* emb_w   = (const float*)d_in[1];
  const float* emb_b   = (const float*)d_in[2];
  const float* norm_w  = (const float*)d_in[3];
  const float* wq      = (const float*)d_in[4];
  const float* wk      = (const float*)d_in[5];
  const float* wv      = (const float*)d_in[6];
  const float* wo      = (const float*)d_in[7];
  const float* w1      = (const float*)d_in[8];
  const float* w2      = (const float*)d_in[9];
  const float* rel_b   = (const float*)d_in[10];
  const float* norm_ow = (const float*)d_in[11];
  const float* cls_w   = (const float*)d_in[12];
  const float* cls_b   = (const float*)d_in[13];
  float* out = (float*)d_out;

  // workspace layout (floats)
  const long NTD = (long)TOK_ * D_;           // 1,310,720
  float* ws = (float*)d_ws;
  float* X    = ws;            // activations
  float* Hb   = X  + NTD;      // rmsnorm output
  float* Q    = Hb + NTD;
  float* Kb   = Q  + NTD;
  float* V    = Kb + NTD;
  float* O    = V  + NTD;
  float* XF   = O  + NTD;
  float* REG  = XF + NTD;      // aliased region: scores | (u, act)
  float* SC   = REG;                         // B*H*S*S = 10,485,760
  float* U    = REG;                         // TOK*W1O = 5,242,880
  float* FACT = REG + (long)TOK_ * W1O_;     // TOK*DI  = 2,621,440

  // 1) embedding
  {
    long n = NTD;
    embed_kernel<<<(unsigned)((n + 255) / 256), 256, 0, stream>>>(latents, emb_w, emb_b, X);
  }

  const float isq = 0.125f; // 1/sqrt(DK)

  for (int i = 0; i < L_; ++i) {
    const float* wq_i = wq + (long)i * D_ * D_;
    const float* wk_i = wk + (long)i * D_ * D_;
    const float* wv_i = wv + (long)i * D_ * D_;
    const float* wo_i = wo + (long)i * D_ * D_;
    const float* w1_i = w1 + (long)i * D_ * W1O_;
    const float* w2_i = w2 + (long)i * DI_ * D_;

    // rmsnorm
    rmsnorm_kernel<<<TOK_, 256, 0, stream>>>(X, norm_w + (long)i * D_, Hb);

    // q/k/v = h @ W  (M=1024,N=1280,K=1280)
    float* qkv_out[3] = {Q, Kb, V};
    const float* qkv_w[3] = {wq_i, wk_i, wv_i};
    for (int j = 0; j < 3; ++j) {
      GemmParams p = {};
      p.M = TOK_; p.N = D_; p.K = D_;
      p.A = Hb; p.lda = D_; p.sAo = 0; p.sAi = 0;
      p.Bm = qkv_w[j]; p.ldb = D_; p.sBo = 0; p.sBi = 0; p.bT = 0;
      p.C = qkv_out[j]; p.ldc = D_; p.sCo = 0; p.sCi = 0;
      p.alpha = 1.0f; p.zdiv = 1;
      run_gemm(p, 1, stream);
    }

    // scores = Q @ K^T / 8   per (b,h): M=S,N=S,K=DK
    {
      GemmParams p = {};
      p.M = S_; p.N = S_; p.K = DK_;
      p.A = Q;  p.lda = D_; p.sAo = (long)S_ * D_; p.sAi = DK_;
      p.Bm = Kb; p.ldb = D_; p.sBo = (long)S_ * D_; p.sBi = DK_; p.bT = 1;
      p.C = SC; p.ldc = S_; p.sCo = (long)H_ * S_ * S_; p.sCi = (long)S_ * S_;
      p.alpha = isq; p.zdiv = H_;
      run_gemm(p, B_ * H_, stream);
    }

    // softmax (+ relative position bias on layer 0)
    softmax_kernel<<<B_ * H_ * S_, 256, 0, stream>>>(SC, rel_b, i == 0 ? 1 : 0);

    // O = attn @ V  per (b,h): M=S,N=DK,K=S
    {
      GemmParams p = {};
      p.M = S_; p.N = DK_; p.K = S_;
      p.A = SC; p.lda = S_; p.sAo = (long)H_ * S_ * S_; p.sAi = (long)S_ * S_;
      p.Bm = V;  p.ldb = D_; p.sBo = (long)S_ * D_; p.sBi = DK_; p.bT = 0;
      p.C = O;  p.ldc = D_; p.sCo = (long)S_ * D_; p.sCi = DK_;
      p.alpha = 1.0f; p.zdiv = H_;
      run_gemm(p, B_ * H_, stream);
    }

    // x = x + O @ Wo
    {
      GemmParams p = {};
      p.M = TOK_; p.N = D_; p.K = D_;
      p.A = O; p.lda = D_;
      p.Bm = wo_i; p.ldb = D_; p.bT = 0;
      p.C = X; p.ldc = D_;
      p.resid = X;
      p.alpha = 1.0f; p.zdiv = 1;
      run_gemm(p, 1, stream);
    }

    // u = x @ W1  (M=1024,N=5120,K=1280)
    {
      GemmParams p = {};
      p.M = TOK_; p.N = W1O_; p.K = D_;
      p.A = X; p.lda = D_;
      p.Bm = w1_i; p.ldb = W1O_; p.bT = 0;
      p.C = U; p.ldc = W1O_;
      p.alpha = 1.0f; p.zdiv = 1;
      run_gemm(p, 1, stream);
    }

    // f = a * gelu(g)
    {
      long n = (long)TOK_ * DI_;
      act_kernel<<<(unsigned)((n + 255) / 256), 256, 0, stream>>>(U, FACT);
    }

    // x = x + f @ W2  (M=1024,N=1280,K=2560)
    {
      GemmParams p = {};
      p.M = TOK_; p.N = D_; p.K = DI_;
      p.A = FACT; p.lda = DI_;
      p.Bm = w2_i; p.ldb = D_; p.bT = 0;
      p.C = X; p.ldc = D_;
      p.resid = X;
      p.alpha = 1.0f; p.zdiv = 1;
      run_gemm(p, 1, stream);
    }
  }

  // final norm
  rmsnorm_kernel<<<TOK_, 256, 0, stream>>>(X, norm_ow, XF);

  // classifier: per b, M=S, N=VOCAB*NPRED, K=D; permuted output + bias
  {
    GemmParams p = {};
    p.M = S_; p.N = VOCAB_ * NPRED_; p.K = D_;
    p.A = XF; p.lda = D_; p.sAo = (long)S_ * D_; p.sAi = 0;
    p.Bm = cls_w; p.ldb = D_; p.sBo = 0; p.sBi = 0; p.bT = 1;
    p.C = out; p.ldc = 0; p.sCo = (long)VOCAB_ * S_ * NPRED_; p.sCi = 0;
    p.bias = cls_b;
    p.alpha = 1.0f; p.zdiv = 1;
    p.cls_mode = 1; p.cls_snp = S_ * NPRED_;
    run_gemm(p, B_, stream);
  }
}
